// MultiHeadAttentionTopK_50362786513339
// MI455X (gfx1250) — matrix operand order repaired and checked by reading, not compiled, not run on hardware
//
#include <hip/hip_runtime.h>
#include <hip/hip_bf16.h>

#define EMBED   1024
#define NHEADS  16
#define HD      64
#define TOPK_N  64
#define BSZ     2
#define SEQ     2048

typedef __bf16 bf16_t;
typedef __attribute__((ext_vector_type(16))) __bf16 v16bf;
typedef __attribute__((ext_vector_type(8)))  __bf16 v8bf;
typedef __attribute__((ext_vector_type(4)))  __bf16 v4bf;
typedef __attribute__((ext_vector_type(8)))  float  v8f;

// ---- WMMA fragment builders (contiguous k order, same for A and B) -----------------
// A (16x32 bf16): lane -> M=lane&15, K = 16*(lane>=16) + e.
__device__ __forceinline__ v16bf frag_A_lds(const bf16_t* base, int ldk, int lane) {
  const int m    = lane & 15;
  const int koff = (lane >> 4) << 4;            // 0 or 16
  const bf16_t* row = base + m * ldk;
  v8bf lo = *(const v8bf*)(row + koff);
  v8bf hi = *(const v8bf*)(row + 8 + koff);
  v16bf r;
#pragma unroll
  for (int i = 0; i < 8; ++i) { r[i] = lo[i]; r[i + 8] = hi[i]; }
  return r;
}
// B (32x16 bf16): lane = N (mod 16), K = 16*(lane>=16) + e. Source stored [N][K].
__device__ __forceinline__ v16bf frag_B_lds(const bf16_t* base, int ldk, int lane) {
  const int n    = lane & 15;
  const int koff = (lane >> 4) << 4;            // 0 or 16
  const bf16_t* p = base + n * ldk + koff;
  v8bf lo = *(const v8bf*)(p);
  v8bf hi = *(const v8bf*)(p + 8);
  v16bf r;
#pragma unroll
  for (int i = 0; i < 8; ++i) { r[i] = lo[i]; r[i + 8] = hi[i]; }
  return r;
}
__device__ __forceinline__ v16bf frag_B_glb(const bf16_t* p) {
  v8bf lo = *(const v8bf*)(p);
  v8bf hi = *(const v8bf*)(p + 8);
  v16bf r;
#pragma unroll
  for (int i = 0; i < 8; ++i) { r[i] = lo[i]; r[i + 8] = hi[i]; }
  return r;
}

// ---- GEMM: Y = X @ W^T + b  (W is [N,K] row-major) -------------------------
// 128x64 tile, 256 threads (8 waves), k-step 32, LDS ping-pong double buffer.
#define GTM 128
#define GTN 64
#define GTK 32
#define LDK 40   // padded row stride (elements); 80B rows keep 16B alignment
#define ABUF (GTM * LDK)
#define BBUF (GTN * LDK)

template<bool IN_BF16, bool OUT_F32>
__global__ __launch_bounds__(256) void gemm_wmma_kernel(
    const void* __restrict__ Xv, const float* __restrict__ W,
    const float* __restrict__ bias, void* __restrict__ Yv,
    int M, int N, int K, int Lseq)
{
  __shared__ __align__(16) bf16_t As[2 * ABUF];
  __shared__ __align__(16) bf16_t Bs[2 * BBUF];
  const int tid  = threadIdx.x;
  const int lane = tid & 31;
  const int wv   = tid >> 5;           // 8 waves, one 16-row strip each
  const int m0   = blockIdx.y * GTM;
  const int n0   = blockIdx.x * GTN;

  const float*  Xf = (const float*)Xv;
  const bf16_t* Xb = (const bf16_t*)Xv;

  float4 xr[4];      // fp32 X staging: 128x32 = 1024 float4 / 256 thr = 4 each
  v8bf   xrb[2];     // bf16 X staging: 128x32 bf16 = 512x16B / 256 thr = 2 each
  float4 wr[2];      // W staging:       64x32 = 512 float4 / 256 thr = 2 each

  auto load_regs = [&](int ks) {
    const int k0 = ks * GTK;
    if constexpr (IN_BF16) {
#pragma unroll
      for (int i = 0; i < 2; ++i) {
        int idx = tid + i * 256;             // 4 16B-chunks per row
        int r = idx >> 2, c8 = (idx & 3) << 3;
        xrb[i] = *(const v8bf*)(Xb + (size_t)(m0 + r) * K + k0 + c8);
      }
    } else {
#pragma unroll
      for (int i = 0; i < 4; ++i) {
        int idx = tid + i * 256;             // 8 float4 per row
        int r = idx >> 3, c4 = (idx & 7) << 2;
        xr[i] = *(const float4*)(Xf + (size_t)(m0 + r) * K + k0 + c4);
      }
    }
#pragma unroll
    for (int i = 0; i < 2; ++i) {
      int idx = tid + i * 256;
      int r = idx >> 3, c4 = (idx & 7) << 2;
      wr[i] = *(const float4*)(W + (size_t)(n0 + r) * K + k0 + c4);
    }
  };

  auto store_lds = [&](int buf) {
    bf16_t* Ab = As + buf * ABUF;
    bf16_t* Bb = Bs + buf * BBUF;
    if constexpr (IN_BF16) {
#pragma unroll
      for (int i = 0; i < 2; ++i) {
        int idx = tid + i * 256;
        int r = idx >> 2, c8 = (idx & 3) << 3;
        *(v8bf*)(Ab + r * LDK + c8) = xrb[i];
      }
    } else {
#pragma unroll
      for (int i = 0; i < 4; ++i) {
        int idx = tid + i * 256;
        int r = idx >> 3, c4 = (idx & 7) << 2;
        v4bf p;
        p[0] = (bf16_t)xr[i].x; p[1] = (bf16_t)xr[i].y;
        p[2] = (bf16_t)xr[i].z; p[3] = (bf16_t)xr[i].w;
        *(v4bf*)(Ab + r * LDK + c4) = p;
      }
    }
#pragma unroll
    for (int i = 0; i < 2; ++i) {
      int idx = tid + i * 256;
      int r = idx >> 3, c4 = (idx & 7) << 2;
      v4bf p;
      p[0] = (bf16_t)wr[i].x; p[1] = (bf16_t)wr[i].y;
      p[2] = (bf16_t)wr[i].z; p[3] = (bf16_t)wr[i].w;
      *(v4bf*)(Bb + r * LDK + c4) = p;
    }
  };

  v8f acc[4] = {};
  const int nk = K / GTK;

  load_regs(0);
  store_lds(0);
  __syncthreads();

  for (int ks = 0; ks < nk; ++ks) {
    const bool more = (ks + 1) < nk;
    if (more) load_regs(ks + 1);               // overlap global with WMMA
    const int cur = ks & 1;
    v16bf a = frag_A_lds(As + cur * ABUF + wv * 16 * LDK, LDK, lane);
#pragma unroll
    for (int nt = 0; nt < 4; ++nt) {
      v16bf b = frag_B_lds(Bs + cur * BBUF + nt * 16 * LDK, LDK, lane);
      acc[nt] = __builtin_amdgcn_wmma_f32_16x16x32_bf16(
          false, a, false, b, (short)0, acc[nt], false, false);
    }
    if (more) store_lds(cur ^ 1);              // other buffer: no race, 1 barrier/iter
    __syncthreads();
  }

  // C/D layout: VGPR r -> row = r + 8*(lane>=16), col = lane&15
  const int col16 = lane & 15;
  const int rbase = (lane >> 4) << 3;
#pragma unroll
  for (int nt = 0; nt < 4; ++nt) {
    int n = n0 + nt * 16 + col16;
    float bval = bias[n];
#pragma unroll
    for (int r = 0; r < 8; ++r) {
      int m = m0 + wv * 16 + rbase + r;
      float y = acc[nt][r] + bval;
      if (OUT_F32) {
        ((float*)Yv)[(size_t)m * N + n] = y;
      } else { // head-major bf16 [B,H,L,hd]
        int bb = m / Lseq, q = m - bb * Lseq;
        int h = n >> 6, d = n & 63;
        ((bf16_t*)Yv)[(((size_t)(bb * NHEADS + h)) * Lseq + q) * HD + d] = (bf16_t)y;
      }
    }
  }
}

// ---- scores (WMMA) + top-64 + softmax + weighted V gather ------------------
#define SROW 2064   // 2048 + 16 pad floats: kills stride-256B bank conflicts

__global__ __launch_bounds__(256) void attn_topk_kernel(
    const bf16_t* __restrict__ Qbf, const bf16_t* __restrict__ Kbf,
    const bf16_t* __restrict__ Vbf, float* __restrict__ attn_out,
    bf16_t* __restrict__ attended)
{
  __shared__ __align__(16) float  S[16 * SROW];   // 132 KB (CDNA5: 320 KB/WGP)
  __shared__ __align__(16) bf16_t Qs[16 * HD];
  __shared__ float topv[16][TOPK_N];
  __shared__ int   topi[16][TOPK_N];

  const int blk = blockIdx.x;          // bh*128 + qtile
  const int qt  = blk & 127;
  const int bh  = blk >> 7;            // b*NHEADS + h
  const int q0  = qt * 16;
  const int tid = threadIdx.x, lane = tid & 31, wv = tid >> 5;

  const bf16_t* Qbase = Qbf + ((size_t)bh * SEQ + q0) * HD;
  const bf16_t* Kbase = Kbf + (size_t)bh * SEQ * HD;
  const bf16_t* Vbase = Vbf + (size_t)bh * SEQ * HD;

  for (int i = tid; i < 16 * HD; i += 256) Qs[i] = Qbase[i];
  __syncthreads();

  v16bf a0 = frag_A_lds(Qs, HD, lane);        // d = 0..31
  v16bf a1 = frag_A_lds(Qs + 32, HD, lane);   // d = 32..63

  // 128 key tiles over 8 waves; B frags straight from global (32B/lane, 128B rows)
  for (int nt = wv; nt < SEQ / 16; nt += 8) {
    const bf16_t* Krow = Kbase + (size_t)(nt * 16 + (lane & 15)) * HD + ((lane >> 4) << 4);
    v16bf b0 = frag_B_glb(Krow);
    v16bf b1 = frag_B_glb(Krow + 32);
    v8f c = {};
    c = __builtin_amdgcn_wmma_f32_16x16x32_bf16(false, a0, false, b0, (short)0, c, false, false);
    c = __builtin_amdgcn_wmma_f32_16x16x32_bf16(false, a1, false, b1, (short)0, c, false, false);
    const int col = nt * 16 + (lane & 15);
    const int rb  = (lane >> 4) << 3;
#pragma unroll
    for (int r = 0; r < 8; ++r)
      S[(rb + r) * SROW + col] = c[r] * 0.125f;   // 1/sqrt(64)
  }
  __syncthreads();

  // top-64 per row: one wave per row (rows wv and wv+8), iterated argmax.
  const float NEG = -__builtin_inff();
  for (int rr = 0; rr < 2; ++rr) {
    const int row = wv + rr * 8;
    float* Srow = S + row * SROW;
    float lmax = NEG; int lidx = lane;
    for (int i = 0; i < SEQ / 32; ++i) {          // lane-strided, conflict-free
      int j = lane + (i << 5);
      float v = Srow[j];
      if (v > lmax) { lmax = v; lidx = j; }
    }
    for (int t = 0; t < TOPK_N; ++t) {
      float m = lmax; int mi = lidx;
#pragma unroll
      for (int off = 16; off >= 1; off >>= 1) {   // argmax, ties -> lowest idx
        float om = __shfl_xor(m, off, 32);
        int   oi = __shfl_xor(mi, off, 32);
        if (om > m || (om == m && oi < mi)) { m = om; mi = oi; }
      }
      if (lane == 0) { topv[row][t] = m; topi[row][t] = mi; }
      if ((mi & 31) == lane) {                    // winner invalidates + rescans
        Srow[mi] = NEG;
        lmax = NEG; lidx = lane;
        for (int i = 0; i < SEQ / 32; ++i) {
          int j = lane + (i << 5);
          float v = Srow[j];
          if (v > lmax) { lmax = v; lidx = j; }
        }
      }
    }
  }

  // softmax (values already sorted desc -> max = [0]) + attn out + V gather.
  for (int rr = 0; rr < 2; ++rr) {
    const int row = wv + rr * 8;     // same wave that produced topv/topi[row]
    const int q = q0 + row;
    float mx = topv[row][0];
    float e0 = __expf(topv[row][lane] - mx);
    float e1 = __expf(topv[row][lane + 32] - mx);
    float s = e0 + e1;
#pragma unroll
    for (int off = 16; off >= 1; off >>= 1) s += __shfl_xor(s, off, 32);
    float inv = 1.0f / s;
    float p0 = e0 * inv, p1 = e1 * inv;
    topv[row][lane] = p0; topv[row][lane + 32] = p1;
    size_t aoff = ((size_t)bh * SEQ + q) * TOPK_N;
    attn_out[aoff + lane]      = p0;
    attn_out[aoff + lane + 32] = p1;

    float acc0 = 0.f, acc1 = 0.f;                 // lane covers d and d+32
#pragma unroll 4
    for (int j = 0; j < TOPK_N; ++j) {
      float w = topv[row][j];
      const bf16_t* vp = Vbase + (size_t)topi[row][j] * HD;
      acc0 += w * (float)vp[lane];
      acc1 += w * (float)vp[lane + 32];
    }
    const int h  = bh & (NHEADS - 1);
    const int bb = bh >> 4;
    bf16_t* op = attended + ((size_t)bb * SEQ + q) * EMBED + h * HD;
    op[lane]      = (bf16_t)acc0;
    op[lane + 32] = (bf16_t)acc1;
  }
}

extern "C" void kernel_launch(void* const* d_in, const int* in_sizes, int n_in,
                              void* d_out, int out_size, void* d_ws, size_t ws_size,
                              hipStream_t stream) {
  const float* local_feat  = (const float*)d_in[0];
  const float* global_feat = (const float*)d_in[1];
  const float* Wq = (const float*)d_in[2];
  const float* bq = (const float*)d_in[3];
  const float* Wk = (const float*)d_in[4];
  const float* bk = (const float*)d_in[5];
  const float* Wv = (const float*)d_in[6];
  const float* bv = (const float*)d_in[7];
  const float* Wo = (const float*)d_in[8];
  const float* bo = (const float*)d_in[9];

  const size_t NT = (size_t)BSZ * SEQ * EMBED;  // 4,194,304
  bf16_t* Qbf = (bf16_t*)d_ws;                  // 8 MB each
  bf16_t* Kbf = Qbf + NT;
  bf16_t* Vbf = Kbf + NT;
  bf16_t* Att = Vbf + NT;

  float* out  = (float*)d_out;                  // [B,LQ,D] f32
  float* attn = out + NT;                       // [B,H,LQ,64] f32

  dim3 grid(EMBED / GTN, (BSZ * SEQ) / GTM);    // 16 x 32
  gemm_wmma_kernel<false, false><<<grid, 256, 0, stream>>>(local_feat,  Wq, bq, Qbf, BSZ*SEQ, EMBED, EMBED, SEQ);
  gemm_wmma_kernel<false, false><<<grid, 256, 0, stream>>>(global_feat, Wk, bk, Kbf, BSZ*SEQ, EMBED, EMBED, SEQ);
  gemm_wmma_kernel<false, false><<<grid, 256, 0, stream>>>(global_feat, Wv, bv, Vbf, BSZ*SEQ, EMBED, EMBED, SEQ);

  attn_topk_kernel<<<BSZ * NHEADS * (SEQ / 16), 256, 0, stream>>>(Qbf, Kbf, Vbf, attn, Att);

  gemm_wmma_kernel<true, true><<<grid, 256, 0, stream>>>(Att, Wo, bo, out, BSZ*SEQ, EMBED, EMBED, SEQ);
}
